// SW_Transformer_Encoder_67568425501415
// MI455X (gfx1250) — matrix operand order repaired and checked by reading, not compiled, not run on hardware
//
#include <hip/hip_runtime.h>

typedef __attribute__((ext_vector_type(16))) _Float16 v16h;
typedef __attribute__((ext_vector_type(8)))  _Float16 v8h;
typedef __attribute__((ext_vector_type(8)))  float    v8f;
typedef __attribute__((ext_vector_type(4)))  unsigned int u32x4;
typedef __attribute__((ext_vector_type(8)))  int i32x8;
typedef __attribute__((ext_vector_type(4)))  int i32x4;

#define WIN_N 343
#define NPAD  352
#define DIM   192
#define NH    6
#define HD    32
#define BATCH 128
#define SCALE 0.17677669529663689f

#if __has_builtin(__builtin_amdgcn_tensor_load_to_lds)
#define HAVE_TDM 1
#endif
#if __has_builtin(__builtin_amdgcn_global_load_async_to_lds_b128)
#define HAVE_ASYNC 1
#endif

__device__ __forceinline__ void wait_tensorcnt0() {
#if __has_builtin(__builtin_amdgcn_s_wait_tensorcnt)
  __builtin_amdgcn_s_wait_tensorcnt(0);
#else
  asm volatile("s_wait_tensorcnt 0x0" ::: "memory");
#endif
}
__device__ __forceinline__ void wait_asynccnt0() {
#if __has_builtin(__builtin_amdgcn_s_wait_asynccnt)
  __builtin_amdgcn_s_wait_asynccnt(0);
#else
  asm volatile("s_wait_asynccnt 0x0" ::: "memory");
#endif
}

__device__ __forceinline__ v8f wmma32f16(v16h a, v16h b, v8f c) {
  // D = A(16x32 f16) * B(32x16 f16) + C(16x16 f32)
  return __builtin_amdgcn_wmma_f32_16x16x32_f16(false, a, false, b, (short)0, c, false, false);
}
__device__ __forceinline__ v8h cvt8(const float* p) {
  v8h r;
#pragma unroll
  for (int i = 0; i < 8; ++i) r[i] = (_Float16)p[i];
  return r;
}
__device__ __forceinline__ v16h cvt16(const float* p) {
  v16h r;
#pragma unroll
  for (int i = 0; i < 16; ++i) r[i] = (_Float16)p[i];
  return r;
}
__device__ __forceinline__ v16h cat(v8h a, v8h b) {
  return __builtin_shufflevector(a, b, 0,1,2,3,4,5,6,7,8,9,10,11,12,13,14,15);
}
// reductions across the 16 lanes sharing one C-matrix row group (wave32)
__device__ __forceinline__ float rmax16(float v) {
  v = fmaxf(v, __shfl_xor(v, 1, 32));
  v = fmaxf(v, __shfl_xor(v, 2, 32));
  v = fmaxf(v, __shfl_xor(v, 4, 32));
  v = fmaxf(v, __shfl_xor(v, 8, 32));
  return v;
}
__device__ __forceinline__ float rsum16(float v) {
  v += __shfl_xor(v, 1, 32);
  v += __shfl_xor(v, 2, 32);
  v += __shfl_xor(v, 4, 32);
  v += __shfl_xor(v, 8, 32);
  return v;
}

// ---- Kernel 0: gather relative-position bias into padded [NH][NPAD][NPAD] f32 (batch-invariant) ----
__global__ __launch_bounds__(256) void bias_gather(const float* __restrict__ rpb,
                                                   const int* __restrict__ ridx,
                                                   float* __restrict__ biasBuf) {
  int i = blockIdx.x * 256 + threadIdx.x;
  const int TOT = NH * NPAD * NPAD;
  if (i >= TOT) return;
  int h = i / (NPAD * NPAD);
  int r = i % (NPAD * NPAD);
  int m = r / NPAD;
  int k = r % NPAD;
  float v;
  if (k >= WIN_N)       v = -1e30f;                       // mask padded key columns
  else if (m >= WIN_N)  v = 0.f;                          // padded query rows: don't care
  else                  v = rpb[ridx[m * WIN_N + k] * NH + h];
  biasBuf[i] = v;
}

// ---- Kernel 1: fused QKV projection for both modalities, f32 -> f16 WMMA, scale folded into Q ----
__global__ __launch_bounds__(256) void qkv_gemm(const float* __restrict__ x0,
                                                const float* __restrict__ x1,
                                                const float* __restrict__ w,
                                                const float* __restrict__ bvec,
                                                _Float16* __restrict__ q0, _Float16* __restrict__ k0b, _Float16* __restrict__ v0b,
                                                _Float16* __restrict__ q1, _Float16* __restrict__ k1b, _Float16* __restrict__ v1b) {
  const int wave = threadIdx.x >> 5, lane = threadIdx.x & 31;
  const int ln = lane & 15, hl = lane >> 4;
  const int mt = blockIdx.x * 8 + wave;   // 2816 row tiles of padded M = 45056
  const int ot = blockIdx.y;              // 36 col tiles of 576
  const int z  = blockIdx.z;
  const float* xin = z ? x1 : x0;
  _Float16* oq = z ? q1 : q0;
  _Float16* ok = z ? k1b : k0b;
  _Float16* ov = z ? v1b : v0b;

  const int m = mt * 16 + ln;
  const int b = m / NPAD, t = m % NPAD;
  const int tc = t < WIN_N ? t : WIN_N - 1;               // clamp pad rows (overwritten below)
  const float* xrow = xin + ((size_t)b * WIN_N + tc) * DIM;
  const int o = ot * 16 + ln;
  const float* wrow = w + (size_t)o * DIM;

  v8f acc = {};
#pragma unroll
  for (int kb = 0; kb < DIM; kb += 32) {
    v16h a  = cvt16(xrow + kb + 16 * hl);
    v16h bm = cvt16(wrow + kb + 16 * hl);
    acc = wmma32f16(a, bm, acc);
  }
#pragma unroll
  for (int r = 0; r < 8; ++r) {
    int m2 = mt * 16 + r + 8 * hl;
    int b2 = m2 / NPAD, t2 = m2 % NPAD;
    float val = acc[r] + bvec[o];
    if (o < DIM) val *= SCALE;              // Q gets the softmax scale
    if (t2 >= WIN_N) val = 0.f;             // zero padded tokens
    int three = o / DIM;
    int hh = (o % DIM) >> 5;
    int hd = o & 31;
    _Float16* dst = three == 0 ? oq : (three == 1 ? ok : ov);
    dst[(((size_t)b2 * NH + hh) * NPAD + t2) * HD + hd] = (_Float16)val;
  }
}

// ---- Kernel 2: flash attention per (batch, head, direction); K staged by TDM, V transposed into LDS ----
__global__ __launch_bounds__(256) void attn(const _Float16* __restrict__ qx, const _Float16* __restrict__ kx, const _Float16* __restrict__ vx,
                                            const _Float16* __restrict__ q1, const _Float16* __restrict__ k1, const _Float16* __restrict__ v1,
                                            const float* __restrict__ biasBuf,
                                            _Float16* __restrict__ a0, _Float16* __restrict__ a1) {
  __shared__ _Float16 lK[NPAD * HD];     // K row-major  [token][dim]
  __shared__ _Float16 lVT[HD * NPAD];    // V transposed [dim][token]
  __shared__ _Float16 lP[8 * 16 * 32];   // per-wave P transpose scratch

  const int tid = threadIdx.x;
  const int bh = blockIdx.x;
  const int dir = blockIdx.z;            // 0: q(x) with k,v(x_1); 1: q(x_1) with k,v(x)
  const int b = bh / NH, h = bh % NH;
  const size_t off = ((size_t)b * NH + h) * NPAD * HD;
  const _Float16* Qs = (dir ? q1 : qx) + off;
  const _Float16* Ks = (dir ? kx : k1) + off;
  const _Float16* Vs = (dir ? vx : v1) + off;
  _Float16* Od = dir ? a1 : a0;

  const int wave = tid >> 5, lane = tid & 31;

  // ---- stage K [352][32] f16 into LDS ----
#ifdef HAVE_TDM
  if (wave == 0) {
    // Tensor DMA descriptor: 2D tensor, data_size=2B, tile 32x352 contiguous.
    unsigned int loff = (unsigned int)(size_t)(__attribute__((address_space(3))) _Float16*)lK;
    unsigned long long ga = (unsigned long long)(size_t)Ks;
    u32x4 g0 = {};
    g0[0] = 1u;                                     // count=1 (valid), user mode, no gather
    g0[1] = loff;                                   // lds_addr
    g0[2] = (unsigned int)ga;                       // global_addr[31:0]
    g0[3] = (unsigned int)(ga >> 32) | (2u << 30);  // global_addr[56:32] | type=2 (image)
    i32x8 g1 = {};
    g1[0] = 1 << 16;                                // data_size=1 (2 bytes), wg_mask=0
    g1[1] = HD << 16;                               // tensor_dim0 = 32 (bits 63:48)
    g1[2] = NPAD << 16;                             // tensor_dim1 = 352 (bits 95:80)
    g1[3] = HD << 16;                               // tile_dim0 = 32 (bits 127:112)
    g1[4] = NPAD;                                   // tile_dim1 = 352 (bits 143:128)
    g1[5] = HD;                                     // tensor_dim0_stride = 32 (bits 191:160)
    i32x4 gz = {};
    i32x8 gz8 = {};
    __builtin_amdgcn_tensor_load_to_lds(g0, g1, gz, gz, gz8, 0);
  }
#elif defined(HAVE_ASYNC)
  for (int i = tid; i < NPAD * HD / 8; i += 256)   // 1408 x 16B lane-transfers
    __builtin_amdgcn_global_load_async_to_lds_b128(
        (__attribute__((address_space(1))) void*)(Ks + (size_t)i * 8),
        (__attribute__((address_space(3))) void*)(lK + (size_t)i * 8), 0, 0);
#else
  for (int i = tid; i < NPAD * HD / 4; i += 256)
    ((uint2*)lK)[i] = ((const uint2*)Ks)[i];
#endif

  // ---- transpose V into LDS [dim][token] (overlaps with K DMA) ----
  for (int i = tid; i < NPAD * HD; i += 256) {
    int t = i >> 5, d = i & 31;
    lVT[d * NPAD + t] = Vs[i];
  }

#ifdef HAVE_TDM
  if (wave == 0) wait_tensorcnt0();
#elif defined(HAVE_ASYNC)
  wait_asynccnt0();
#endif
  __syncthreads();

  const int ln = lane & 15, hl = lane >> 4;
  _Float16* Pw = &lP[wave * 512];

  for (int qt = wave; qt < NPAD / 16; qt += 8) {
    const _Float16* qp = Qs + (size_t)(qt * 16 + ln) * HD;
    v16h aQ = *(const v16h*)(qp + 16 * hl);
    v8f acc0 = {}, acc1 = {};
    float mrow[8], lrow[8];
#pragma unroll
    for (int r = 0; r < 8; ++r) { mrow[r] = -1e30f; lrow[r] = 0.f; }

    for (int c = 0; c < NPAD / 32; ++c) {
      // scores: S = Q(16x32) * K^T(32x16) x2  (full head dim in one WMMA, K=32)
      const v16h bk0 = *(const v16h*)&lK[(32 * c + ln) * HD + 16 * hl];
      const v16h bk1 = *(const v16h*)&lK[(32 * c + 16 + ln) * HD + 16 * hl];
      v8f zc = {};
      v8f s0 = wmma32f16(aQ, bk0, zc);
      v8f s1 = wmma32f16(aQ, bk1, zc);
      const float* bp = biasBuf + ((size_t)h * NPAD + qt * 16 + 8 * hl) * NPAD + 32 * c + ln;
#pragma unroll
      for (int r = 0; r < 8; ++r) {
        s0[r] += bp[r * NPAD];
        s1[r] += bp[r * NPAD + 16];
      }
      // online softmax (rows striped across 16-lane groups)
#pragma unroll
      for (int r = 0; r < 8; ++r) {
        float tmax = rmax16(fmaxf(s0[r], s1[r]));
        float mn = fmaxf(mrow[r], tmax);
        float corr = __expf(mrow[r] - mn);
        mrow[r] = mn;
        float p0 = __expf(s0[r] - mn);
        float p1 = __expf(s1[r] - mn);
        lrow[r] = lrow[r] * corr + rsum16(p0 + p1);
        acc0[r] *= corr; acc1[r] *= corr;
        s0[r] = p0; s1[r] = p1;
      }
      // transpose P (C-layout -> A-layout) through per-wave LDS; same-wave DS ops are in order
#pragma unroll
      for (int r = 0; r < 8; ++r) {
        int row = r + 8 * hl;
        Pw[row * 32 + ln]      = (_Float16)s0[r];
        Pw[row * 32 + 16 + ln] = (_Float16)s1[r];
      }
      v16h aP = *(const v16h*)&Pw[ln * 32 + 16 * hl];
      // O += P(16x32) * V(32x32) as two 16x16 output halves
      const v16h bv0 = *(const v16h*)&lVT[(size_t)ln * NPAD + 32 * c + 16 * hl];
      const v16h bv1 = *(const v16h*)&lVT[(size_t)(16 + ln) * NPAD + 32 * c + 16 * hl];
      acc0 = wmma32f16(aP, bv0, acc0);
      acc1 = wmma32f16(aP, bv1, acc1);
    }
    _Float16* op = Od + ((size_t)b * NPAD + qt * 16 + 8 * hl) * DIM + h * HD + ln;
#pragma unroll
    for (int r = 0; r < 8; ++r) {
      float inv = 1.0f / lrow[r];
      op[(size_t)r * DIM]      = (_Float16)(acc0[r] * inv);
      op[(size_t)r * DIM + 16] = (_Float16)(acc1[r] * inv);
    }
  }
}

// ---- Kernel 3: output projection, f16 WMMA, f32 epilogue + bias, drop pad rows ----
__global__ __launch_bounds__(256) void proj_gemm(const _Float16* __restrict__ a0,
                                                 const _Float16* __restrict__ a1,
                                                 const float* __restrict__ w,
                                                 const float* __restrict__ bvec,
                                                 float* __restrict__ out) {
  const int wave = threadIdx.x >> 5, lane = threadIdx.x & 31;
  const int ln = lane & 15, hl = lane >> 4;
  const int mt = blockIdx.x * 8 + wave;
  const int ot = blockIdx.y;
  const int dir = blockIdx.z;
  const _Float16* ain = dir ? a1 : a0;
  const _Float16* arow = ain + (size_t)(mt * 16 + ln) * DIM;
  const int o = ot * 16 + ln;
  const float* wrow = w + (size_t)o * DIM;
  v8f acc = {};
#pragma unroll
  for (int kb = 0; kb < DIM; kb += 32) {
    v16h a  = *(const v16h*)(arow + kb + 16 * hl);
    v16h bm = cvt16(wrow + kb + 16 * hl);
    acc = wmma32f16(a, bm, acc);
  }
  float* obase = out + (size_t)dir * BATCH * WIN_N * DIM;
#pragma unroll
  for (int r = 0; r < 8; ++r) {
    int m2 = mt * 16 + r + 8 * hl;
    int b2 = m2 / NPAD, t2 = m2 % NPAD;
    if (t2 < WIN_N)
      obase[((size_t)b2 * WIN_N + t2) * DIM + o] = acc[r] + bvec[o];
  }
}

extern "C" void kernel_launch(void* const* d_in, const int* in_sizes, int n_in,
                              void* d_out, int out_size, void* d_ws, size_t ws_size,
                              hipStream_t stream) {
  (void)in_sizes; (void)n_in; (void)out_size; (void)ws_size;
  const float* x    = (const float*)d_in[0];
  const float* x1   = (const float*)d_in[1];
  const float* qkvw = (const float*)d_in[2];
  const float* qkvb = (const float*)d_in[3];
  const float* pw   = (const float*)d_in[4];
  const float* pb   = (const float*)d_in[5];
  const float* rpb  = (const float*)d_in[6];
  const int*   ridx = (const int*)d_in[7];
  float* out = (float*)d_out;

  // workspace layout: 6 qkv f16 buffers + 2 attn f16 buffers (each B*NPAD*DIM halfs) + bias f32
  const size_t SZH = (size_t)BATCH * NH * NPAD * HD;  // == BATCH*NPAD*DIM = 8,650,752 halfs
  _Float16* ws = (_Float16*)d_ws;
  _Float16* qx = ws;       _Float16* kx = qx + SZH; _Float16* vx = kx + SZH;
  _Float16* q1 = vx + SZH; _Float16* k1 = q1 + SZH; _Float16* v1 = k1 + SZH;
  _Float16* a0 = v1 + SZH; _Float16* a1 = a0 + SZH;
  float* biasBuf = (float*)(a1 + SZH);

  bias_gather<<<(NH * NPAD * NPAD + 255) / 256, 256, 0, stream>>>(rpb, ridx, biasBuf);
  qkv_gemm<<<dim3(352, 36, 2), 256, 0, stream>>>(x, x1, qkvw, qkvb, qx, kx, vx, q1, k1, v1);
  attn<<<dim3(BATCH * NH, 1, 2), 256, 0, stream>>>(qx, kx, vx, q1, k1, v1, biasBuf, a0, a1);
  proj_gemm<<<dim3(352, 12, 2), 256, 0, stream>>>(a0, a1, pw, pb, out);
}
